// ClusterMasking_18588618457558
// MI455X (gfx1250) — compile-verified, matrix operand order checked by reading, not run on hardware
//
#include <hip/hip_runtime.h>
#include <hip/hip_bf16.h>
#include <stdint.h>

// Problem constants (B=64, N=1024, D=768 from the reference)
#define B_  64
#define N_  1024
#define D_  768
#define NUM_ANCHORS 51      // int(1024 * 0.05)
#define ANCH_PAD    64      // padded to 4 WMMA M-tiles of 16
#define SIM_TH      0.75f

typedef __attribute__((ext_vector_type(16))) __bf16 v16bf;
typedef __attribute__((ext_vector_type(8)))  float  v8f;

__device__ __forceinline__ unsigned hash_u32(unsigned x) {
    x ^= x >> 16; x *= 0x7feb352dU;
    x ^= x >> 15; x *= 0x846ca68bU;
    x ^= x >> 16;
    return x;
}

// float -> bf16 round-to-nearest-even (returned in low 16 bits)
__device__ __forceinline__ unsigned f2bf(float f) {
    unsigned u = __float_as_uint(f);
    u += 0x7fffU + ((u >> 16) & 1U);
    return u >> 16;
}

// ---------------------------------------------------------------- zero flags
__global__ void k_zero_flags(unsigned* __restrict__ flags) {
    flags[blockIdx.x * 256 + threadIdx.x] = 0u;
}

// ------------------------------------------------- per-patch normalize -> bf16
// One block (192 threads) per patch row; each thread owns one float4 (b128
// load) and emits 4 packed bf16 (one b64 store). Single reduction produces
// mean, unbiased var AND the L2 norm of the normalized vector:
//   ssd = sum x^2 - (sum x)^2 / D ;  var = ssd/(D-1);  inv = rsqrt(var+1e-6)
//   |normed|^2 = ssd * inv^2  ->  unit_i = (x_i-mean)*inv / max(|normed|,1e-12)
__global__ void __launch_bounds__(192)
k_normalize(const float* __restrict__ patches, unsigned short* __restrict__ unit) {
    const int row = blockIdx.x;                 // b*N + n
    const int tid = threadIdx.x;                // 0..191 (D/4 = 192 float4)
    const float4* p4 = reinterpret_cast<const float4*>(patches + (size_t)row * D_);
    const float4 x = p4[tid];

    float s = x.x + x.y + x.z + x.w;
    float q = x.x * x.x + x.y * x.y + x.z * x.z + x.w * x.w;

    __shared__ float sh1[192];
    __shared__ float sh2[192];
    sh1[tid] = s; sh2[tid] = q;
    __syncthreads();
    if (tid < 64) { sh1[tid] += sh1[tid + 128]; sh2[tid] += sh2[tid + 128]; }
    __syncthreads();
    for (int o = 64; o > 0; o >>= 1) {
        if (tid < o) { sh1[tid] += sh1[tid + o]; sh2[tid] += sh2[tid + o]; }
        __syncthreads();
    }
    const float sum  = sh1[0];
    const float sq   = sh2[0];
    const float mean = sum * (1.0f / D_);
    const float ssd  = sq - sum * mean;
    const float var  = ssd * (1.0f / (D_ - 1));
    const float inv  = rsqrtf(var + 1e-6f);
    float l2 = sqrtf(fmaxf(ssd, 0.0f) * inv * inv);
    l2 = fmaxf(l2, 1e-12f);
    const float scale = inv / l2;

    uint2 pk;
    pk.x = f2bf((x.x - mean) * scale) | (f2bf((x.y - mean) * scale) << 16);
    pk.y = f2bf((x.z - mean) * scale) | (f2bf((x.w - mean) * scale) << 16);
    reinterpret_cast<uint2*>(unit + (size_t)row * D_)[tid] = pk;
}

// -------------------------------------------------------- anchor generation
// Deterministic stand-in for jax.random.permutation: odd stride mod 1024
// gives 51 distinct indices. Slots [51..63] pad with anchor 0 (masked by a
// validity check in the GEMM kernel). Anchors are forced into the mask.
__global__ void k_anchors(int* __restrict__ anchors, unsigned* __restrict__ flags) {
    const int b = blockIdx.x;
    const int i = threadIdx.x;                  // blockDim.x == 64
    const unsigned h    = hash_u32(0x2A0000u ^ ((unsigned)b * 0x9E3779B9u));
    const unsigned off  = h & (N_ - 1);
    const unsigned step = ((h >> 10) | 1u) & (N_ - 1);   // odd -> invertible mod 1024
    const int ii  = (i < NUM_ANCHORS) ? i : 0;
    const int idx = (int)((off + (unsigned)ii * step) & (N_ - 1));
    anchors[b * ANCH_PAD + i] = idx;
    if (i < NUM_ANCHORS) atomicOr(&flags[b * N_ + idx], 1u);
}

// -------------------------------- bf16 WMMA similarity GEMM + threshold/OR
// One wave owns one N-tile of 16 columns and iterates ALL 4 anchor M-tiles in
// registers: per K-step it loads the B fragment once (8 VGPRs) plus 4 A
// fragments and issues 4 back-to-back v_wmma_f32_16x16x32_bf16 into 4
// accumulators. 4x less B traffic than one-wave-per-M-tile, no LDS/barriers.
// Grid: B*16 blocks x 128 threads (4 waves, wave w -> ntile base+w).
// A fragment (16x32 bf16): lane holds row = lane%16; low lanes K[0..7,16..23],
// high lanes K[8..15,24..31] -> two b128 loads/lane. B = unit^T, so a B
// column is a contiguous unit row: identical load pattern.
__global__ void __launch_bounds__(128)
k_sim_cluster(const unsigned short* __restrict__ unit,
              const int* __restrict__ anchors,
              unsigned* __restrict__ flags) {
    const int blk   = blockIdx.x;
    const int b     = blk >> 4;                       // batch
    const int ntile = ((blk & 15) << 2) + (threadIdx.x >> 5);
    const int lane  = threadIdx.x & 31;
    const int ml    = lane & 15;
    const int hi    = lane >> 4;         // 0: K-low half, 1: K-high half
    const int lo    = hi * 8;

    const int col = ntile * 16 + ml;
    const unsigned short* baseB = unit + ((size_t)(b * N_ + col)) * D_;

    const unsigned short* baseA[4];
    #pragma unroll
    for (int m = 0; m < 4; ++m) {
        const int arow = anchors[b * ANCH_PAD + m * 16 + ml];
        baseA[m] = unit + ((size_t)(b * N_ + arow)) * D_;
    }

    v8f c[4] = {v8f{}, v8f{}, v8f{}, v8f{}};
    #pragma unroll 2
    for (int k0 = 0; k0 < D_; k0 += 32) {
        union { uint4 u[2]; v16bf v; } fb;
        fb.u[0] = *reinterpret_cast<const uint4*>(baseB + k0 + lo);
        fb.u[1] = *reinterpret_cast<const uint4*>(baseB + k0 + 16 + lo);
        #pragma unroll
        for (int m = 0; m < 4; ++m) {
            union { uint4 u[2]; v16bf v; } fa;
            fa.u[0] = *reinterpret_cast<const uint4*>(baseA[m] + k0 + lo);
            fa.u[1] = *reinterpret_cast<const uint4*>(baseA[m] + k0 + 16 + lo);
            c[m] = __builtin_amdgcn_wmma_f32_16x16x32_bf16(
                       /*neg_a=*/false, fa.v, /*neg_b=*/false, fb.v,
                       /*c_mod=*/(short)0, c[m], /*reuse_a=*/false, /*reuse_b=*/false);
        }
    }

    // C layout: VGPR r -> M=r (lanes 0-15) / M=r+8 (lanes 16-31), N = lane%16
    bool hit = false;
    #pragma unroll
    for (int m = 0; m < 4; ++m) {
        #pragma unroll
        for (int r = 0; r < 8; ++r) {
            const int slot = m * 16 + r + hi * 8;
            hit = hit || ((slot < NUM_ANCHORS) && (c[m][r] > SIM_TH));
        }
    }
    if (hit) atomicOr(&flags[b * N_ + col], 1u);
}

// --------------------------------------------- top-up mask to MIN_MASK_RATIO
// need = floor((0.5 - total/1024)*1024) = 512 - total (exact). Extras chosen
// in a deterministic pseudo-random order over unmasked positions.
__global__ void __launch_bounds__(256)
k_extra(unsigned* __restrict__ flags) {
    const int b   = blockIdx.x;
    const int tid = threadIdx.x;
    __shared__ int sh[256];
    int cnt = 0;
    for (int i = tid; i < N_; i += 256) cnt += (flags[b * N_ + i] ? 1 : 0);
    sh[tid] = cnt;
    __syncthreads();
    for (int o = 128; o > 0; o >>= 1) {
        if (tid < o) sh[tid] += sh[tid + o];
        __syncthreads();
    }
    if (tid == 0) {
        const int total = sh[0];
        int need = (total < N_ / 2) ? (N_ / 2 - total) : 0;
        if (need > 0) {
            const unsigned h    = hash_u32(0x51u ^ ((unsigned)b * 0x85ebca6bU));
            const unsigned off  = h & (N_ - 1);
            const unsigned step = ((h >> 9) | 1u) & (N_ - 1);
            for (int j = 0; j < N_ && need > 0; ++j) {
                const unsigned pos = (off + (unsigned)j * step) & (N_ - 1);
                unsigned* f = &flags[b * N_ + pos];
                if (*f == 0u) { *f = 1u; --need; }
            }
        }
    }
}

// -------------------------------------------------- apply mask, emit outputs
// 192 threads/block, one float4 (b128) load + store per thread.
__global__ void __launch_bounds__(192)
k_apply(const float* __restrict__ patches, const unsigned* __restrict__ flags,
        float* __restrict__ out, float* __restrict__ outmask) {
    const int row = blockIdx.x;                 // b*N + n
    const unsigned m = flags[row];
    const float keep = m ? 0.0f : 1.0f;
    const float4* p4 = reinterpret_cast<const float4*>(patches + (size_t)row * D_);
    float4* o4 = reinterpret_cast<float4*>(out + (size_t)row * D_);
    float4 v = p4[threadIdx.x];
    v.x *= keep; v.y *= keep; v.z *= keep; v.w *= keep;
    o4[threadIdx.x] = v;
    if (threadIdx.x == 0) outmask[row] = m ? 1.0f : 0.0f;
}

extern "C" void kernel_launch(void* const* d_in, const int* in_sizes, int n_in,
                              void* d_out, int out_size, void* d_ws, size_t ws_size,
                              hipStream_t stream) {
    (void)in_sizes; (void)n_in; (void)out_size; (void)ws_size;

    const float* patches = (const float*)d_in[0];
    float* out     = (float*)d_out;                       // (B,N,D) masked patches
    float* outmask = out + (size_t)B_ * N_ * D_;          // (B,N) mask as 0/1 float

    // workspace layout: unit bf16 (96 MB) | flags (256 KB) | anchors (16 KB)
    char* ws = (char*)d_ws;
    unsigned short* unit = (unsigned short*)ws;
    unsigned* flags = (unsigned*)(ws + (size_t)B_ * N_ * D_ * sizeof(unsigned short));
    int* anchors = (int*)((char*)flags + (size_t)B_ * N_ * sizeof(unsigned));

    k_zero_flags <<< (B_ * N_) / 256, 256, 0, stream >>> (flags);
    k_normalize  <<< B_ * N_,          192, 0, stream >>> (patches, unit);
    k_anchors    <<< B_,                64, 0, stream >>> (anchors, flags);
    k_sim_cluster<<< B_ * 16,          128, 0, stream >>> (unit, anchors, flags);
    k_extra      <<< B_,               256, 0, stream >>> (flags);
    k_apply      <<< B_ * N_,          192, 0, stream >>> (patches, flags, out, outmask);
}
